// DeepEMD_73254962201130
// MI455X (gfx1250) — matrix operand order repaired and checked by reading, not compiled, not run on hardware
//
#include <hip/hip_runtime.h>
#include <math.h>

typedef __attribute__((ext_vector_type(16))) _Float16 v16h;
typedef __attribute__((ext_vector_type(8)))  float    v8f;

#define NN    64
#define CCH   1024
#define TEMP  12.5f
#define EPSR  0.05f
#define NIT   100

union HFrag { unsigned int u[8]; v16h v; };

__global__ __launch_bounds__(256)
void deepemd_kernel(const float* __restrict__ proto,
                    const float* __restrict__ query,
                    float* __restrict__ out) {
  // Double-buffered f16 staging tiles, NODE-major so WMMA A/B fragment slots are
  // single aligned b32 LDS loads (row stride 68B = odd dword -> conflict-free).
  __shared__ _Float16 Qs[2][64][34];
  __shared__ _Float16 Ps[2][64][34];
  __shared__ float simS[64][65];
  __shared__ float Km[64][65];
  __shared__ float red[8][6 * 64];     // per-wave partials of 6 per-node stats
  __shared__ float chmP[64], invP[64], chmQ[64], invQ[64];
  __shared__ float rr[64], ccm[64], uu[64], vv[64];
  __shared__ float red256[256];
  __shared__ float red64[64];
  __shared__ float scal[2];

  const int  b    = blockIdx.x;
  const int  t    = threadIdx.x;
  const int  wave = t >> 5;
  const int  lane = t & 31;
  const long base = (long)b * CCH * NN;
  const float* P = proto + base;
  const float* Q = query + base;

  // ---------------- Pass 1: streaming stats (one read of both tensors) --------------
  // wave w handles channels c = w, w+8, ...; lane holds nodes {2*lane, 2*lane+1}.
  float2 csP = {0.f, 0.f}, sqP = {0.f, 0.f};
  float2 csQ = {0.f, 0.f}, sqQ = {0.f, 0.f};
  float2 w1a = {0.f, 0.f}, w2a = {0.f, 0.f};
  for (int it = 0; it < CCH / 8; ++it) {
    const int c = wave + 8 * it;
    const float2 p2 = *(const float2*)(P + c * NN + 2 * lane);
    const float2 q2 = *(const float2*)(Q + c * NN + 2 * lane);
    if (it + 8 < CCH / 8) {            // stream 16KB (64 channels) ahead per tensor
      __builtin_prefetch(P + (c + 64) * NN + 2 * lane, 0, 3);
      __builtin_prefetch(Q + (c + 64) * NN + 2 * lane, 0, 3);
    }
    float sp = p2.x + p2.y;
    float sq = q2.x + q2.y;
    #pragma unroll
    for (int m = 16; m >= 1; m >>= 1) {
      sp += __shfl_xor(sp, m, 32);
      sq += __shfl_xor(sq, m, 32);
    }
    const float poolP = sp * (1.f / NN);   // mean over spatial of proto channel c
    const float poolQ = sq * (1.f / NN);
    csP.x += p2.x;          csP.y += p2.y;
    sqP.x += p2.x * p2.x;   sqP.y += p2.y * p2.y;
    csQ.x += q2.x;          csQ.y += q2.y;
    sqQ.x += q2.x * q2.x;   sqQ.y += q2.y * q2.y;
    w1a.x += q2.x * poolP;  w1a.y += q2.y * poolP;   // weight_1 = <query, pooled proto>
    w2a.x += p2.x * poolQ;  w2a.y += p2.y * poolQ;   // weight_2 = <proto, pooled query>
  }
  {
    const int h0 = 2 * lane;
    red[wave][0 * 64 + h0] = csP.x;  red[wave][0 * 64 + h0 + 1] = csP.y;
    red[wave][1 * 64 + h0] = sqP.x;  red[wave][1 * 64 + h0 + 1] = sqP.y;
    red[wave][2 * 64 + h0] = csQ.x;  red[wave][2 * 64 + h0 + 1] = csQ.y;
    red[wave][3 * 64 + h0] = sqQ.x;  red[wave][3 * 64 + h0 + 1] = sqQ.y;
    red[wave][4 * 64 + h0] = w1a.x;  red[wave][4 * 64 + h0 + 1] = w1a.y;
    red[wave][5 * 64 + h0] = w2a.x;  red[wave][5 * 64 + h0 + 1] = w2a.y;
  }
  __syncthreads();
  if (t < 64) {
    float s[6];
    #pragma unroll
    for (int k = 0; k < 6; ++k) {
      float a = 0.f;
      #pragma unroll
      for (int w = 0; w < 8; ++w) a += red[w][k * 64 + t];
      s[k] = a;
    }
    // centering mean + inverse node-norm (||x - m||^2 = sum x^2 - (sum x)^2 / C)
    chmP[t] = s[0] * (1.f / CCH);
    chmQ[t] = s[2] * (1.f / CCH);
    const float vP = fmaxf(s[1] - s[0] * s[0] * (1.f / CCH), 0.f);
    const float vQ = fmaxf(s[3] - s[2] * s[2] * (1.f / CCH), 0.f);
    invP[t] = 1.f / fmaxf(sqrtf(vP), 1e-8f);
    invQ[t] = 1.f / fmaxf(sqrtf(vQ), 1e-8f);
    rr[t]  = fmaxf(s[4], 0.f) + 0.001f;   // weight_1 (query nodes -> rows)
    ccm[t] = fmaxf(s[5], 0.f) + 0.001f;   // weight_2 (proto nodes -> cols)
  }
  __syncthreads();
  if (t == 0) {
    float a = 0.f, c2 = 0.f;
    for (int i = 0; i < 64; ++i) { a += rr[i]; c2 += ccm[i]; }
    scal[0] = 1.f / a;
    scal[1] = 1.f / c2;
  }
  __syncthreads();
  if (t < 64) { rr[t] *= scal[0]; ccm[t] *= scal[1]; vv[t] = 1.f; }
  __syncthreads();

  // ---------------- Pass 2: sim = Qn^T * Pn via v_wmma_f32_16x16x32_f16 -------------
  // 16 output tiles (4x4 of 16x16); wave w owns (qi = w>>1, pi = 2*(w&1) + {0,1}).
  // Software-pipelined: global loads for slab ks+1 issue before the WMMAs of ks,
  // then land in the alternate LDS buffer; one barrier per slab.
  v8f acc0 = {}; v8f acc1 = {};
  const int qi  = wave >> 1;
  const int pb0 = (wave & 1) * 32;
  const int mr  = lane & 15;
  const int kh  = lane >> 4;
  const int clS = t >> 4;               // staged channels: clS and clS+16
  const int hwS = (t & 15) * 4;         // staged node base

  float4 pv0, qv0, pv1, qv1;
  // preload + stage slab 0 into buffer 0
  pv0 = *(const float4*)(P + (long)clS        * NN + hwS);
  qv0 = *(const float4*)(Q + (long)clS        * NN + hwS);
  pv1 = *(const float4*)(P + (long)(clS + 16) * NN + hwS);
  qv1 = *(const float4*)(Q + (long)(clS + 16) * NN + hwS);
  #pragma unroll
  for (int j = 0; j < 4; ++j) {
    const float mP = chmP[hwS + j], iP = invP[hwS + j];
    const float mQ = chmQ[hwS + j], iQ = invQ[hwS + j];
    const float pe0 = (j == 0 ? pv0.x : j == 1 ? pv0.y : j == 2 ? pv0.z : pv0.w);
    const float qe0 = (j == 0 ? qv0.x : j == 1 ? qv0.y : j == 2 ? qv0.z : qv0.w);
    const float pe1 = (j == 0 ? pv1.x : j == 1 ? pv1.y : j == 2 ? pv1.z : pv1.w);
    const float qe1 = (j == 0 ? qv1.x : j == 1 ? qv1.y : j == 2 ? qv1.z : qv1.w);
    Ps[0][hwS + j][clS]      = (_Float16)((pe0 - mP) * iP);
    Qs[0][hwS + j][clS]      = (_Float16)((qe0 - mQ) * iQ);
    Ps[0][hwS + j][clS + 16] = (_Float16)((pe1 - mP) * iP);
    Qs[0][hwS + j][clS + 16] = (_Float16)((qe1 - mQ) * iQ);
  }
  __syncthreads();

  for (int ks = 0; ks < CCH / 32; ++ks) {
    const int cur = ks & 1;
    const bool more = (ks + 1 < CCH / 32);
    if (more) {                         // issue next slab's loads early
      const long c0n = (long)(ks + 1) * 32;
      pv0 = *(const float4*)(P + (c0n + clS)      * NN + hwS);
      qv0 = *(const float4*)(Q + (c0n + clS)      * NN + hwS);
      pv1 = *(const float4*)(P + (c0n + clS + 16) * NN + hwS);
      qv1 = *(const float4*)(Q + (c0n + clS + 16) * NN + hwS);
    }
    HFrag A, B0, B1;
    #pragma unroll
    for (int v = 0; v < 8; ++v) {
      // A (16x32 f16): VGPR v holds K pair (v<4 ? 2v : 16+2(v-4)) + 8*(lane>>4)
      const int ka = (v < 4 ? 2 * v : 16 + 2 * (v - 4)) + 8 * kh;
      A.u[v]  = *(const unsigned int*)&Qs[cur][qi * 16 + mr][ka];
      // B (32x16 f16): VGPR v holds K pair (v<4 ? 2v : 16+2(v-4)) + 8*(lane>>4), N = lane&15
      const int kb = (v < 4 ? 2 * v : 16 + 2 * (v - 4)) + 8 * kh;
      B0.u[v] = *(const unsigned int*)&Ps[cur][pb0 + mr][kb];
      B1.u[v] = *(const unsigned int*)&Ps[cur][pb0 + 16 + mr][kb];
    }
    acc0 = __builtin_amdgcn_wmma_f32_16x16x32_f16(false, A.v, false, B0.v,
                                                  (short)0, acc0, false, false);
    acc1 = __builtin_amdgcn_wmma_f32_16x16x32_f16(false, A.v, false, B1.v,
                                                  (short)0, acc1, false, false);
    if (more) {                         // stage next slab into the other buffer
      const int nb = cur ^ 1;
      #pragma unroll
      for (int j = 0; j < 4; ++j) {
        const float mP = chmP[hwS + j], iP = invP[hwS + j];
        const float mQ = chmQ[hwS + j], iQ = invQ[hwS + j];
        const float pe0 = (j == 0 ? pv0.x : j == 1 ? pv0.y : j == 2 ? pv0.z : pv0.w);
        const float qe0 = (j == 0 ? qv0.x : j == 1 ? qv0.y : j == 2 ? qv0.z : qv0.w);
        const float pe1 = (j == 0 ? pv1.x : j == 1 ? pv1.y : j == 2 ? pv1.z : pv1.w);
        const float qe1 = (j == 0 ? qv1.x : j == 1 ? qv1.y : j == 2 ? qv1.z : qv1.w);
        Ps[nb][hwS + j][clS]      = (_Float16)((pe0 - mP) * iP);
        Qs[nb][hwS + j][clS]      = (_Float16)((qe0 - mQ) * iQ);
        Ps[nb][hwS + j][clS + 16] = (_Float16)((pe1 - mP) * iP);
        Qs[nb][hwS + j][clS + 16] = (_Float16)((qe1 - mQ) * iQ);
      }
    }
    __syncthreads();
  }
  #pragma unroll
  for (int r2 = 0; r2 < 8; ++r2) {    // C/D layout: m = r + 8*(lane>>4), n = lane&15
    simS[qi * 16 + r2 + 8 * kh][pb0 + mr]      = acc0[r2];
    simS[qi * 16 + r2 + 8 * kh][pb0 + 16 + mr] = acc1[r2];
  }
  __syncthreads();

  // ---------------- Sinkhorn: K = exp(-(1-sim)/eps), 100 iterations -----------------
  for (int i = t; i < 64 * 64; i += 256) {
    const int r2 = i >> 6, c2 = i & 63;
    Km[r2][c2] = __expf((simS[r2][c2] - 1.f) * (1.f / EPSR));
  }
  __syncthreads();
  const int ri  = t >> 2;
  const int seg = (t & 3) * 16;
  for (int itn = 0; itn < NIT; ++itn) {
    float s = 0.f;
    #pragma unroll
    for (int j = 0; j < 16; ++j) s += Km[ri][seg + j] * vv[seg + j];
    red256[t] = s;
    __syncthreads();
    if (t < 64)
      uu[t] = rr[t] / fmaxf(red256[4 * t] + red256[4 * t + 1] +
                            red256[4 * t + 2] + red256[4 * t + 3], 1e-30f);
    __syncthreads();
    s = 0.f;
    #pragma unroll
    for (int i2 = 0; i2 < 16; ++i2) s += Km[seg + i2][ri] * uu[seg + i2];
    red256[t] = s;
    __syncthreads();
    if (t < 64)
      vv[t] = ccm[t] / fmaxf(red256[4 * t] + red256[4 * t + 1] +
                             red256[4 * t + 2] + red256[4 * t + 3], 1e-30f);
    __syncthreads();
  }
  // final u with the last v
  {
    float s = 0.f;
    #pragma unroll
    for (int j = 0; j < 16; ++j) s += Km[ri][seg + j] * vv[seg + j];
    red256[t] = s;
    __syncthreads();
    if (t < 64)
      uu[t] = rr[t] / fmaxf(red256[4 * t] + red256[4 * t + 1] +
                            red256[4 * t + 2] + red256[4 * t + 3], 1e-30f);
    __syncthreads();
  }

  // ---------------- logits = sum(sim * (u_i K_ij v_j)) * T/N ------------------------
  {
    float s = 0.f;
    const float ui = uu[ri];
    #pragma unroll
    for (int j = 0; j < 16; ++j)
      s += simS[ri][seg + j] * (ui * Km[ri][seg + j] * vv[seg + j]);
    red256[t] = s;
    __syncthreads();
    if (t < 64)
      red64[t] = red256[4 * t] + red256[4 * t + 1] + red256[4 * t + 2] + red256[4 * t + 3];
    __syncthreads();
    if (t == 0) {
      float tot = 0.f;
      for (int i = 0; i < 64; ++i) tot += red64[i];
      out[b] = tot * (TEMP / (float)NN);
    }
  }
}

extern "C" void kernel_launch(void* const* d_in, const int* in_sizes, int n_in,
                              void* d_out, int out_size, void* d_ws, size_t ws_size,
                              hipStream_t stream) {
  const float* proto = (const float*)d_in[0];
  const float* query = (const float*)d_in[1];
  float* out = (float*)d_out;
  (void)in_sizes; (void)n_in; (void)out_size; (void)d_ws; (void)ws_size;
  deepemd_kernel<<<512, 256, 0, stream>>>(proto, query, out);
}
